// WanSelfAttention_17051020165625
// MI455X (gfx1250) — compile-verified, matrix operand order checked by reading, not run on hardware
//
#include <hip/hip_runtime.h>

// ---------------------------------------------------------------------------
// Problem constants (B=1): L=4096 tokens, C=2048 channels, H=16 heads, D=128
// ---------------------------------------------------------------------------
#define LSEQ 4096
#define CDIM 2048
#define NHEAD 16
#define DHEAD 128

typedef __bf16 bf16;
typedef __attribute__((ext_vector_type(16))) __bf16 v16bf;
typedef __attribute__((ext_vector_type(8)))  float  v8f;
typedef __attribute__((ext_vector_type(4)))  unsigned int u32x4;
typedef __attribute__((ext_vector_type(4)))  unsigned int v4u;
typedef __attribute__((ext_vector_type(8)))  int v8i;
typedef __attribute__((ext_vector_type(4)))  int v4i;

union FragU { v16bf v; u32x4 u[2]; };

// float <-> bf16 without relying on __bf16 arithmetic support
static __device__ __forceinline__ bf16 f2bf(float f) {
    union { float f; unsigned int u; } x; x.f = f;
    unsigned int r = x.u + 0x7FFFu + ((x.u >> 16) & 1u);   // RNE
    union { unsigned short s; bf16 b; } y;
    y.s = (unsigned short)(r >> 16);
    return y.b;
}

// Load one 16-bit A/B WMMA fragment: per-lane K layout is
//   elem i<8  -> k = half*8 + i          (first 16B chunk)
//   elem i>=8 -> k = 16 + half*8 + (i-8) (second 16B chunk)
static __device__ __forceinline__ v16bf load_frag(const bf16* p_lo, const bf16* p_hi) {
    FragU f;
    f.u[0] = *reinterpret_cast<const u32x4*>(p_lo);
    f.u[1] = *reinterpret_cast<const u32x4*>(p_hi);
    return f.v;
}

static __device__ __forceinline__ v8f wmma_bf16(v16bf a, v16bf b, v8f c) {
    return __builtin_amdgcn_wmma_f32_16x16x32_bf16(
        /*neg_a=*/false, a, /*neg_b=*/false, b,
        /*c_mod=*/(short)0, c, /*reuse_a=*/false, /*reuse_b=*/false);
}

// ---------------------------------------------------------------------------
// Tensor Data Mover: async 2D tile load (bf16) global -> LDS.
//   tile_d0 elems along contiguous dim (stride 1), tile_d1 rows with
//   row stride `stride0` elements. Tracked by TENSORcnt.
// ---------------------------------------------------------------------------
static __device__ __forceinline__ void tdm_load_2d_bf16(
    unsigned lds_byte_addr, const bf16* gptr,
    unsigned tile_d0, unsigned tile_d1, unsigned long long stride0_elems) {
    unsigned long long ga = (unsigned long long)(size_t)gptr;
    v4u g0;
    g0[0] = 1u;                                            // count=1, user D#
    g0[1] = lds_byte_addr;                                 // lds_addr
    g0[2] = (unsigned)(ga & 0xFFFFFFFFu);                  // global_addr[31:0]
    g0[3] = (unsigned)((ga >> 32) & 0x01FFFFFFu) | (2u << 30);  // [56:32]|type=2

    // tensor dims == tile dims (exact tiling, no OOB clamping needed)
    unsigned d0 = tile_d0, d1 = tile_d1;
    unsigned long long s0 = stride0_elems;
    v8i g1;
    g1[0] = (int)(1u << 16);                               // data_size=1 (2B)
    g1[1] = (int)(d0 << 16);                               // tensor_dim0[15:0]
    g1[2] = (int)((d0 >> 16) | (d1 << 16));                // dim0 hi | dim1 lo
    g1[3] = (int)((d1 >> 16) | (tile_d0 << 16));           // dim1 hi | tile_dim0
    g1[4] = (int)(tile_d1 & 0xFFFFu);                      // tile_dim1, tile_dim2=0
    g1[5] = (int)(s0 & 0xFFFFFFFFull);                     // stride0[31:0]
    g1[6] = (int)((s0 >> 32) & 0xFFFFull);                 // stride0[47:32]
    g1[7] = 0;

    v4i z4; z4[0] = 0; z4[1] = 0; z4[2] = 0; z4[3] = 0;
#if defined(__clang_major__) && __clang_major__ >= 23
    v8i z8;
    #pragma unroll
    for (int i = 0; i < 8; ++i) z8[i] = 0;
    __builtin_amdgcn_tensor_load_to_lds(g0, g1, z4, z4, z8, 0);
#else
    __builtin_amdgcn_tensor_load_to_lds(g0, g1, z4, z4, 0);
#endif
}

// Low 32 bits of a generic pointer to LDS == LDS byte address (flat->LDS map)
static __device__ __forceinline__ unsigned lds_addr_of(const void* p) {
    return (unsigned)(size_t)p;
}

// ---------------------------------------------------------------------------
// 1) f32 -> bf16 conversion
// ---------------------------------------------------------------------------
__global__ __launch_bounds__(256) void cvt_f32_bf16_k(const float* __restrict__ src,
                                                      bf16* __restrict__ dst, int n) {
    int i = blockIdx.x * 256 + threadIdx.x;
    if (i < n) dst[i] = f2bf(src[i]);
}

// ---------------------------------------------------------------------------
// 2) GEMM: out[M,N] = A[M,K](bf16) @ W[N,K](bf16)^T + bias[N], f32 out.
//    Block = 256 thr (8 waves). Block tile = 64 rows x 128 cols.
//    Each wave: 64x16 slab = 4 accumulators; B fragment reused 4x.
//    A panel (64 x 128K) staged in LDS by the Tensor Data Mover,
//    double-buffered: next panel DMA overlaps current panel's 16 WMMAs.
// ---------------------------------------------------------------------------
#define KCHUNK 128
#define MROWS  64
__global__ __launch_bounds__(256) void gemm_bf16_wmma_k(
    const bf16* __restrict__ A, const bf16* __restrict__ W,
    const float* __restrict__ bias, float* __restrict__ out,
    int M, int N, int K) {
    __shared__ __align__(16) bf16 lds[2][MROWS][KCHUNK];

    const int t    = threadIdx.x;
    const int wave = t >> 5;
    const int lane = t & 31;
    const int n16  = lane & 15;      // A-row m / B-col n within 16x16 tile
    const int half = lane >> 4;
    const int m0   = blockIdx.x * MROWS;
    const int n    = blockIdx.y * 128 + wave * 16 + n16;

    v8f acc[4];
    #pragma unroll
    for (int mb = 0; mb < 4; ++mb)
        #pragma unroll
        for (int r = 0; r < 8; ++r) acc[mb][r] = 0.0f;

    const bf16* wrow   = W + (size_t)n * K;
    const int   nchunk = K / KCHUNK;

    if (wave == 0) {   // TDM issue: one DMA per workgroup (EXEC-independent)
        tdm_load_2d_bf16(lds_addr_of(&lds[0][0][0]),
                         A + (size_t)m0 * K, KCHUNK, MROWS, (unsigned long long)K);
    }

    for (int chunk = 0; chunk < nchunk; ++chunk) {
        const int buf = chunk & 1;
        if (wave == 0) {
            if (chunk + 1 < nchunk) {
                tdm_load_2d_bf16(lds_addr_of(&lds[buf ^ 1][0][0]),
                                 A + (size_t)m0 * K + (size_t)(chunk + 1) * KCHUNK,
                                 KCHUNK, MROWS, (unsigned long long)K);
                __builtin_amdgcn_s_wait_tensorcnt(1);   // oldest DMA done
            } else {
                __builtin_amdgcn_s_wait_tensorcnt(0);
            }
        }
        __syncthreads();   // panel `buf` visible to all waves

        const int k0 = chunk * KCHUNK;
        #pragma unroll
        for (int kk = 0; kk < 4; ++kk) {
            v16bf bfr = load_frag(wrow + k0 + kk * 32 + half * 8,
                                  wrow + k0 + kk * 32 + 16 + half * 8);
            #pragma unroll
            for (int mb = 0; mb < 4; ++mb) {
                const bf16* arow = &lds[buf][mb * 16 + n16][kk * 32];
                v16bf af = load_frag(arow + half * 8, arow + 16 + half * 8);
                acc[mb] = wmma_bf16(af, bfr, acc[mb]);
            }
        }
        __syncthreads();   // all reads done before buffer is overwritten
    }

    const float bv = bias[n];
    #pragma unroll
    for (int mb = 0; mb < 4; ++mb)
        #pragma unroll
        for (int r = 0; r < 8; ++r)
            out[(size_t)(m0 + mb * 16 + r + half * 8) * N + n] = acc[mb][r] + bv;
}

// ---------------------------------------------------------------------------
// 3) Fused RMSNorm(q,k) + RoPE(q,k) + repack:
//    Q,K -> bf16 [H][L][D]; V -> bf16 transposed [H][D][L].
// ---------------------------------------------------------------------------
__global__ __launch_bounds__(256) void norm_rope_pack_k(
    const float* __restrict__ qf, const float* __restrict__ kf,
    const float* __restrict__ vf,
    const float* __restrict__ gq, const float* __restrict__ gk,
    const float* __restrict__ cosT, const float* __restrict__ sinT,
    bf16* __restrict__ Qb, bf16* __restrict__ Kb, bf16* __restrict__ Vtb) {
    const int l  = blockIdx.x;
    const int t  = threadIdx.x;
    const int c0 = t * 8;
    const size_t row = (size_t)l * CDIM;

    float q8[8], k8[8];
    float sq = 0.f, sk = 0.f;
    #pragma unroll
    for (int j = 0; j < 8; ++j) {
        q8[j] = qf[row + c0 + j];
        k8[j] = kf[row + c0 + j];
        sq += q8[j] * q8[j];
        sk += k8[j] * k8[j];
    }
    #pragma unroll
    for (int o = 16; o > 0; o >>= 1) {
        sq += __shfl_xor(sq, o, 32);
        sk += __shfl_xor(sk, o, 32);
    }
    __shared__ float rq[8], rk[8];
    if ((t & 31) == 0) { rq[t >> 5] = sq; rk[t >> 5] = sk; }
    __syncthreads();
    float tq = 0.f, tk = 0.f;
    #pragma unroll
    for (int w = 0; w < 8; ++w) { tq += rq[w]; tk += rk[w]; }
    const float rmsq = rsqrtf(tq / (float)CDIM + 1e-6f);
    const float rmsk = rsqrtf(tk / (float)CDIM + 1e-6f);

    #pragma unroll
    for (int j = 0; j < 8; ++j) {
        q8[j] *= rmsq * gq[c0 + j];
        k8[j] *= rmsk * gk[c0 + j];
    }

    const int h     = c0 >> 7;
    const int dbase = c0 & 127;
    #pragma unroll
    for (int p = 0; p < 4; ++p) {
        const int d0 = dbase + 2 * p;
        const float cs = cosT[(size_t)l * DHEAD + d0];
        const float sn = sinT[(size_t)l * DHEAD + d0 + 1];
        float q1 = q8[2 * p], q2 = q8[2 * p + 1];
        float k1 = k8[2 * p], k2 = k8[2 * p + 1];
        q8[2 * p]     = q1 * cs - q2 * sn;
        q8[2 * p + 1] = q1 * sn + q2 * cs;
        k8[2 * p]     = k1 * cs - k2 * sn;
        k8[2 * p + 1] = k1 * sn + k2 * cs;
    }

    bf16* qdst = Qb + ((size_t)h * LSEQ + l) * DHEAD + dbase;
    bf16* kdst = Kb + ((size_t)h * LSEQ + l) * DHEAD + dbase;
    #pragma unroll
    for (int j = 0; j < 8; ++j) {
        qdst[j] = f2bf(q8[j]);
        kdst[j] = f2bf(k8[j]);
        Vtb[((size_t)h * DHEAD + dbase + j) * LSEQ + l] = f2bf(vf[row + c0 + j]);
    }
}

// ---------------------------------------------------------------------------
// 4) Flash attention, one wave per (head, 16-query tile).
//    S^T = K x Q^T  (C-layout: lane = query, rows = keys)
//    O^T = V^T x P^T (C-layout: lane = query, rows = d)
//    P^T B-fragment built lane-locally from the S^T accumulators.
// ---------------------------------------------------------------------------
__global__ __launch_bounds__(32) void flash_attn_k(
    const bf16* __restrict__ Q, const bf16* __restrict__ Kk,
    const bf16* __restrict__ Vt, bf16* __restrict__ attnO) {
    const int lane = threadIdx.x;
    const int n16  = lane & 15;
    const int half = lane >> 4;
    const int q0   = blockIdx.x * 16;
    const int h    = blockIdx.y;

    const bf16* Qh = Q  + (size_t)h * LSEQ * DHEAD;
    const bf16* Kh = Kk + (size_t)h * LSEQ * DHEAD;
    const bf16* Vh = Vt + (size_t)h * DHEAD * LSEQ;   // [D][L]

    v16bf qfr[4];
    const bf16* qrow = Qh + (size_t)(q0 + n16) * DHEAD;
    #pragma unroll
    for (int s = 0; s < 4; ++s)
        qfr[s] = load_frag(qrow + s * 32 + half * 8, qrow + s * 32 + 16 + half * 8);

    v8f ot[8];
    #pragma unroll
    for (int ts = 0; ts < 8; ++ts)
        #pragma unroll
        for (int r = 0; r < 8; ++r) ot[ts][r] = 0.0f;

    float m_i = -3.0e38f, l_i = 0.0f;
    const float scale = 0.08838834764831845f;   // 1/sqrt(128)

    for (int kb = 0; kb < LSEQ; kb += 32) {
        v8f s0, s1;
        #pragma unroll
        for (int r = 0; r < 8; ++r) { s0[r] = 0.0f; s1[r] = 0.0f; }

        const bf16* kr0 = Kh + (size_t)(kb + n16) * DHEAD;
        const bf16* kr1 = Kh + (size_t)(kb + 16 + n16) * DHEAD;
        #pragma unroll
        for (int s = 0; s < 4; ++s) {
            v16bf ka0 = load_frag(kr0 + s * 32 + half * 8, kr0 + s * 32 + 16 + half * 8);
            v16bf ka1 = load_frag(kr1 + s * 32 + half * 8, kr1 + s * 32 + 16 + half * 8);
            s0 = wmma_bf16(ka0, qfr[s], s0);
            s1 = wmma_bf16(ka1, qfr[s], s1);
        }

        float mx = -3.0e38f;
        #pragma unroll
        for (int r = 0; r < 8; ++r) {
            s0[r] *= scale; s1[r] *= scale;
            mx = fmaxf(mx, fmaxf(s0[r], s1[r]));
        }
        mx = fmaxf(mx, __shfl_xor(mx, 16, 32));
        const float m_new = fmaxf(m_i, mx);
        const float corr  = __expf(m_i - m_new);

        float p0[8], p1[8];
        float rs = 0.0f;
        #pragma unroll
        for (int r = 0; r < 8; ++r) {
            p0[r] = __expf(s0[r] - m_new);
            p1[r] = __expf(s1[r] - m_new);
            rs += p0[r] + p1[r];
        }
        rs += __shfl_xor(rs, 16, 32);
        l_i = l_i * corr + rs;
        m_i = m_new;

        FragU pf;
        #pragma unroll
        for (int r = 0; r < 8; ++r) {
            pf.v[r]     = f2bf(p0[r]);
            pf.v[8 + r] = f2bf(p1[r]);
        }

        #pragma unroll
        for (int ts = 0; ts < 8; ++ts) {
            const bf16* vrow = Vh + (size_t)(ts * 16 + n16) * LSEQ + kb;
            v16bf va = load_frag(vrow + half * 8, vrow + 16 + half * 8);
            #pragma unroll
            for (int r = 0; r < 8; ++r) ot[ts][r] *= corr;
            ot[ts] = wmma_bf16(va, pf.v, ot[ts]);
        }
    }

    const float inv = 1.0f / l_i;
    #pragma unroll
    for (int ts = 0; ts < 8; ++ts) {
        #pragma unroll
        for (int r = 0; r < 8; ++r) {
            const int d = ts * 16 + r + half * 8;
            attnO[(size_t)(q0 + n16) * CDIM + h * DHEAD + d] = f2bf(ot[ts][r] * inv);
        }
    }
}

// ---------------------------------------------------------------------------
// Host launch
// ---------------------------------------------------------------------------
extern "C" void kernel_launch(void* const* d_in, const int* in_sizes, int n_in,
                              void* d_out, int out_size, void* d_ws, size_t ws_size,
                              hipStream_t stream) {
    (void)in_sizes; (void)n_in; (void)out_size; (void)ws_size;

    const float* x  = (const float*)d_in[0];
    const float* fc = (const float*)d_in[1];
    const float* fs = (const float*)d_in[2];
    const float* Wq = (const float*)d_in[3];
    const float* bq = (const float*)d_in[4];
    const float* Wk = (const float*)d_in[5];
    const float* bk = (const float*)d_in[6];
    const float* Wv = (const float*)d_in[7];
    const float* bv = (const float*)d_in[8];
    const float* Wo = (const float*)d_in[9];
    const float* bo = (const float*)d_in[10];
    const float* gq = (const float*)d_in[11];
    const float* gk = (const float*)d_in[12];

    char* ws = (char*)d_ws;
    const size_t MB = 1024 * 1024;
    bf16*  xb    = (bf16*)(ws + 0);            // 16 MB  x bf16 [L][C]
    bf16*  wb    = (bf16*)(ws + 16 * MB);      //  8 MB  shared weight slot
    float* qf    = (float*)(ws + 24 * MB);     // 32 MB  q f32 [L][C]
    float* kf    = (float*)(ws + 56 * MB);     // 32 MB
    float* vf    = (float*)(ws + 88 * MB);     // 32 MB
    bf16*  Qb    = (bf16*)(ws + 120 * MB);     // 16 MB  [H][L][D]
    bf16*  Kb    = (bf16*)(ws + 136 * MB);     // 16 MB  [H][L][D]
    bf16*  Vtb   = (bf16*)(ws + 152 * MB);     // 16 MB  [H][D][L]
    bf16*  attnb = (bf16*)(ws + 24 * MB);      // reuse qf slot (qf dead by then)

    const int nX = LSEQ * CDIM;
    const int nW = CDIM * CDIM;
    dim3 b256(256);
    dim3 gX((nX + 255) / 256), gW((nW + 255) / 256);
    dim3 gGemm(LSEQ / MROWS, CDIM / 128);

    cvt_f32_bf16_k<<<gX, b256, 0, stream>>>(x, xb, nX);

    cvt_f32_bf16_k<<<gW, b256, 0, stream>>>(Wq, wb, nW);
    gemm_bf16_wmma_k<<<gGemm, b256, 0, stream>>>(xb, wb, bq, qf, LSEQ, CDIM, CDIM);

    cvt_f32_bf16_k<<<gW, b256, 0, stream>>>(Wk, wb, nW);
    gemm_bf16_wmma_k<<<gGemm, b256, 0, stream>>>(xb, wb, bk, kf, LSEQ, CDIM, CDIM);

    cvt_f32_bf16_k<<<gW, b256, 0, stream>>>(Wv, wb, nW);
    gemm_bf16_wmma_k<<<gGemm, b256, 0, stream>>>(xb, wb, bv, vf, LSEQ, CDIM, CDIM);

    norm_rope_pack_k<<<LSEQ, b256, 0, stream>>>(qf, kf, vf, gq, gk, fc, fs, Qb, Kb, Vtb);

    flash_attn_k<<<dim3(LSEQ / 16, NHEAD), dim3(32), 0, stream>>>(Qb, Kb, Vtb, attnb);

    cvt_f32_bf16_k<<<gW, b256, 0, stream>>>(Wo, wb, nW);
    gemm_bf16_wmma_k<<<gGemm, b256, 0, stream>>>(attnb, wb, bo, (float*)d_out,
                                                 LSEQ, CDIM, CDIM);
}
